// MultiHeadAttention_704374636760
// MI455X (gfx1250) — matrix operand order repaired and checked by reading, not compiled, not run on hardware
//
#include <hip/hip_runtime.h>
#include <hip/hip_bf16.h>

typedef _Float16 f16;
typedef __attribute__((ext_vector_type(16))) _Float16 v16h;
typedef __attribute__((ext_vector_type(8)))  _Float16 v8h;
typedef __attribute__((ext_vector_type(8)))  float    v8f;

#define B_  2
#define S_  2048
#define D_  1024
#define H_  16
#define HD_ 64
#define M_  (B_*S_)

__device__ inline v8f wmma32(v16h a, v16h b, v8f c) {
  return __builtin_amdgcn_wmma_f32_16x16x32_f16(false, a, false, b, (short)0, c,
                                                false, false);
}

// ---- CDNA5 async global->LDS copy (tracked by ASYNCcnt) --------------------
__device__ inline void async_g2l_b128(unsigned int lds_off, const void* gaddr) {
  asm volatile("global_load_async_to_lds_b128 %0, %1, off"
               :: "v"(lds_off), "v"(gaddr) : "memory");
}
__device__ inline void wait_async0() {
  asm volatile("s_wait_asynccnt 0x0" ::: "memory");
}
__device__ inline unsigned int lds_off_u32(const void* p) {
  return (unsigned int)(unsigned long long)p;   // generic LDS ptr low 32 = offset
}

// ---- Fragment loaders (16-bit WMMA operands, wave32; contiguous k = 16*(lane/16) + s order) -------------------
// A (16x32): lanes 0-15 -> rows, K 0..15; lanes 16-31 -> K 16..31.
__device__ inline v16h load_a_f16(const f16* __restrict__ src, int ld,
                                  int row0, int k0, int lane) {
  int m  = row0 + (lane & 15);
  int kb = (lane & 16) ? 16 : 0;
  const f16* p0 = src + (size_t)m * ld + k0 + kb;
  const f16* p1 = p0 + 8;
  v16h a;
#pragma unroll
  for (int j = 0; j < 8; ++j) { a[j] = p0[j]; a[8 + j] = p1[j]; }
  return a;
}

// B (32x16): lane holds column n = lane&15; lanes 0-15 -> K 0..15, 16-31 -> 16..31.
// "T": element (k,n) = src[(n0+n)*ld + k0+k]  (k contiguous: W rows / K rows).
__device__ inline v16h load_bT_f16(const f16* __restrict__ src, int ld,
                                   int n0, int k0, int lane) {
  int n  = n0 + (lane & 15);
  int kb = (lane & 16) ? 16 : 0;
  const f16* p = src + (size_t)n * ld + k0 + kb;
  v16h b;
#pragma unroll
  for (int i = 0; i < 16; ++i) b[i] = p[i];
  return b;
}

// Row-major: element (k,n) = src[(k0+k)*ld + n0+n]  (V tiles, k strided).
__device__ inline v16h load_b_f16(const f16* __restrict__ src, int ld,
                                  int k0, int n0, int lane) {
  int n  = n0 + (lane & 15);
  int kb = (lane & 16) ? 16 : 0;
  const f16* p = src + (size_t)(k0 + kb) * ld + n;
  v16h b;
#pragma unroll
  for (int i = 0; i < 16; ++i) b[i] = p[(size_t)i * ld];
  return b;
}

// ---- width-16 wave reductions ----------------------------------------------
__device__ inline float hmax16(float v) {
#pragma unroll
  for (int m = 1; m < 16; m <<= 1) v = fmaxf(v, __shfl_xor(v, m, 16));
  return v;
}
__device__ inline float hsum16(float v) {
#pragma unroll
  for (int m = 1; m < 16; m <<= 1) v += __shfl_xor(v, m, 16);
  return v;
}

// ---- fp32 -> f16 bulk convert (one-shot; bandwidth bound) ------------------
__global__ __launch_bounds__(256) void cvt_kernel(const float* __restrict__ src,
                                                  f16* __restrict__ dst, int n) {
  int i = (blockIdx.x * 256 + threadIdx.x) * 8;
  if (i + 8 <= n) {
    const float4* s = (const float4*)(src + i);
    float4 x = s[0], y = s[1];
    v8h o;
    o[0] = (f16)x.x; o[1] = (f16)x.y; o[2] = (f16)x.z; o[3] = (f16)x.w;
    o[4] = (f16)y.x; o[5] = (f16)y.y; o[6] = (f16)y.z; o[7] = (f16)y.w;
    *(v8h*)(dst + i) = o;
  }
}

// ---- Projection GEMM (f16 x f16, f32 accumulate) ---------------------------
// Block: 256 thr (8 waves) -> 16 x 512 output; A tile (16x32 f16) async-staged
// into double-buffered LDS; each wave computes 16x64 (4 WMMA / k-step).
__global__ __launch_bounds__(256) void proj_kernel(const f16* __restrict__ X,
                                                   const f16* __restrict__ W,
                                                   const float* __restrict__ bias,
                                                   f16* __restrict__ out) {
  __shared__ __align__(16) f16 aTile[2][16 * 32];
  int tid = threadIdx.x, lane = tid & 31, wave = tid >> 5;
  int m0 = blockIdx.x * 16;
  int n0 = blockIdx.y * 512 + wave * 64;
  unsigned int aB0 = lds_off_u32(aTile[0]);
  unsigned int aB1 = lds_off_u32(aTile[1]);
  int row = tid >> 2, c16 = tid & 3;                    // 64 x 16B chunks
  v8f acc[4] = {};
  if (tid < 64)
    async_g2l_b128(aB0 + (unsigned)(row * 32 + c16 * 8) * 2u,
                   X + (size_t)(m0 + row) * D_ + c16 * 8);
  int cur = 0;
  for (int k0 = 0; k0 < D_; k0 += 32) {
    wait_async0();
    __syncthreads();                                    // aTile[cur] published
    if (k0 + 32 < D_ && tid < 64)                       // prefetch next tile
      async_g2l_b128((cur ? aB0 : aB1) + (unsigned)(row * 32 + c16 * 8) * 2u,
                     X + (size_t)(m0 + row) * D_ + (k0 + 32) + c16 * 8);
    v16h a = load_a_f16(aTile[cur], 32, 0, 0, lane);
#pragma unroll
    for (int t = 0; t < 4; ++t)
      acc[t] = wmma32(a, load_bT_f16(W, D_, n0 + t * 16, k0, lane), acc[t]);
    __syncthreads();                                    // reads done before restage
    cur ^= 1;
  }
  int rbase = (lane & 16) ? 8 : 0;                      // C layout: VGPR r -> row r+rbase
#pragma unroll
  for (int t = 0; t < 4; ++t) {
    int n = n0 + t * 16 + (lane & 15);
    float bn = bias[n];
    int h = n >> 6, hd = n & 63;
#pragma unroll
    for (int r = 0; r < 8; ++r) {
      int m = m0 + rbase + r;
      int b_ = m >> 11, s = m & (S_ - 1);
      out[(((size_t)b_ * H_ + h) * S_ + s) * HD_ + hd] = (f16)(acc[t][r] + bn);
    }
  }
}

// ---- Flash attention -------------------------------------------------------
// Block: 128 thr (4 waves) owns 64 queries of one (b,h); K/V 32x64 f16 tiles
// async-staged into double-buffered LDS, shared by all 4 waves.
__global__ __launch_bounds__(128) void attn_kernel(const f16* __restrict__ Q,
                                                   const f16* __restrict__ K,
                                                   const f16* __restrict__ V,
                                                   f16* __restrict__ O) {
  __shared__ __align__(16) f16 kTile[2][32 * 64];
  __shared__ __align__(16) f16 vTile[2][32 * 64];
  __shared__ __align__(16) f16 pbuf[4][16 * 32];
  int tid = threadIdx.x, lane = tid & 31, wave = tid >> 5;
  int bh = blockIdx.x >> 5;                             // S_/64 == 32 query-blocks
  int q0 = (blockIdx.x & 31) * 64 + wave * 16;
  const f16* Qb = Q + (size_t)bh * S_ * HD_;
  const f16* Kb = K + (size_t)bh * S_ * HD_;
  const f16* Vb = V + (size_t)bh * S_ * HD_;

  v16h qa0 = load_a_f16(Qb, HD_, q0, 0, lane);
  v16h qa1 = load_a_f16(Qb, HD_, q0, 32, lane);

  v8f acc[4] = {};
  float mrow[8], lrow[8];
#pragma unroll
  for (int r = 0; r < 8; ++r) { mrow[r] = -__builtin_inff(); lrow[r] = 0.f; }

  f16* myP = pbuf[wave];
  const float scale = 0.125f;                           // 1/sqrt(64)
  int col = lane & 15;
  int rbase = (lane & 16) ? 8 : 0;
  unsigned int kB0 = lds_off_u32(kTile[0]), kB1 = lds_off_u32(kTile[1]);
  unsigned int vB0 = lds_off_u32(vTile[0]), vB1 = lds_off_u32(vTile[1]);

  auto stage = [&](unsigned int kB, unsigned int vB, int jrow) {
    for (int c = tid; c < 256; c += 128) {              // 32 rows x 128B each tile
      int r_ = c >> 3, c16 = c & 7;
      async_g2l_b128(kB + (unsigned)c * 16u, Kb + (size_t)(jrow + r_) * HD_ + c16 * 8);
      async_g2l_b128(vB + (unsigned)c * 16u, Vb + (size_t)(jrow + r_) * HD_ + c16 * 8);
    }
  };

  stage(kB0, vB0, 0);
  int cur = 0;
  for (int j = 0; j < S_; j += 32) {
    wait_async0();
    __syncthreads();                                    // tiles[cur] published
    if (j + 32 < S_)                                    // prefetch next K/V tiles
      stage(cur ? kB0 : kB1, cur ? vB0 : vB1, j + 32);

    const f16* kT = kTile[cur];
    const f16* vT = vTile[cur];
    // scores: two 16x16 f32 tiles (keys j..j+15, j+16..j+31)
    v8f s0 = {}, s1 = {};
    s0 = wmma32(qa0, load_bT_f16(kT, HD_, 0, 0, lane), s0);
    s0 = wmma32(qa1, load_bT_f16(kT, HD_, 0, 32, lane), s0);
    s1 = wmma32(qa0, load_bT_f16(kT, HD_, 16, 0, lane), s1);
    s1 = wmma32(qa1, load_bT_f16(kT, HD_, 16, 32, lane), s1);

    // online softmax; VGPR r holds row (rbase+r)
#pragma unroll
    for (int r = 0; r < 8; ++r) {
      float v0 = s0[r] * scale, v1 = s1[r] * scale;
      float mx = hmax16(fmaxf(v0, v1));
      float mn = fmaxf(mrow[r], mx);
      float corr = __expf(mrow[r] - mn);
      float p0 = __expf(v0 - mn), p1 = __expf(v1 - mn);
      lrow[r] = lrow[r] * corr + hsum16(p0 + p1);
      mrow[r] = mn;
#pragma unroll
      for (int t = 0; t < 4; ++t) acc[t][r] *= corr;
      s0[r] = p0; s1[r] = p1;
    }

    // C-layout -> A-layout via per-wave LDS round trip
#pragma unroll
    for (int r = 0; r < 8; ++r) {
      myP[(rbase + r) * 32 + col]      = (f16)s0[r];
      myP[(rbase + r) * 32 + col + 16] = (f16)s1[r];
    }
    __syncthreads();
    v16h pf = load_a_f16(myP, 32, 0, 0, lane);
#pragma unroll
    for (int t = 0; t < 4; ++t)
      acc[t] = wmma32(pf, load_b_f16(vT, HD_, 0, t * 16, lane), acc[t]);
    __syncthreads();                                    // reads done before restage
    cur ^= 1;
  }

  int b = bh / H_, h = bh % H_;
#pragma unroll
  for (int r = 0; r < 8; ++r) {
    int m = q0 + rbase + r;
    float inv = 1.f / lrow[r];
    size_t base = (((size_t)b * S_ + m) * H_ + h) * HD_ + col;   // (B,S,D) flat
#pragma unroll
    for (int t = 0; t < 4; ++t) O[base + t * 16] = (f16)(acc[t][r] * inv);
  }
}

// ---- Output projection: d_out[m,n] = A[m,:] . Wo[n,:] + bo[n] (f32 out) ----
__global__ __launch_bounds__(256) void out_proj_kernel(const f16* __restrict__ A,
                                                       const f16* __restrict__ W,
                                                       const float* __restrict__ bias,
                                                       float* __restrict__ out) {
  __shared__ __align__(16) f16 aTile[2][16 * 32];
  int tid = threadIdx.x, lane = tid & 31, wave = tid >> 5;
  int m0 = blockIdx.x * 16;
  int n0 = blockIdx.y * 512 + wave * 64;
  unsigned int aB0 = lds_off_u32(aTile[0]);
  unsigned int aB1 = lds_off_u32(aTile[1]);
  int row = tid >> 2, c16 = tid & 3;
  v8f acc[4] = {};
  if (tid < 64)
    async_g2l_b128(aB0 + (unsigned)(row * 32 + c16 * 8) * 2u,
                   A + (size_t)(m0 + row) * D_ + c16 * 8);
  int cur = 0;
  for (int k0 = 0; k0 < D_; k0 += 32) {
    wait_async0();
    __syncthreads();
    if (k0 + 32 < D_ && tid < 64)
      async_g2l_b128((cur ? aB0 : aB1) + (unsigned)(row * 32 + c16 * 8) * 2u,
                     A + (size_t)(m0 + row) * D_ + (k0 + 32) + c16 * 8);
    v16h a = load_a_f16(aTile[cur], 32, 0, 0, lane);
#pragma unroll
    for (int t = 0; t < 4; ++t)
      acc[t] = wmma32(a, load_bT_f16(W, D_, n0 + t * 16, k0, lane), acc[t]);
    __syncthreads();
    cur ^= 1;
  }
  int rbase = (lane & 16) ? 8 : 0;
#pragma unroll
  for (int t = 0; t < 4; ++t) {
    int n = n0 + t * 16 + (lane & 15);
    float bn = bias[n];
#pragma unroll
    for (int r = 0; r < 8; ++r) {
      int m = m0 + rbase + r;
      out[(size_t)m * D_ + n] = acc[t][r] + bn;
    }
  }
}

extern "C" void kernel_launch(void* const* d_in, const int* in_sizes, int n_in,
                              void* d_out, int out_size, void* d_ws, size_t ws_size,
                              hipStream_t stream) {
  const float* query = (const float*)d_in[0];
  const float* key   = (const float*)d_in[1];
  const float* value = (const float*)d_in[2];
  const float* Wq = (const float*)d_in[3];
  const float* bq = (const float*)d_in[4];
  const float* Wk = (const float*)d_in[5];
  const float* bk = (const float*)d_in[6];
  const float* Wv = (const float*)d_in[7];
  const float* bv = (const float*)d_in[8];
  const float* Wo = (const float*)d_in[9];
  const float* bo = (const float*)d_in[10];

  const size_t MD = (size_t)M_ * D_;                  // 4M elements
  const size_t DD = (size_t)D_ * D_;                  // 1M elements
  f16* q16  = (f16*)d_ws;                             // (B,H,S,Hd)
  f16* k16  = q16 + MD;
  f16* v16  = k16 + MD;
  f16* o16  = v16 + MD;                               // (B,S,D)
  f16* xq16 = o16 + MD;                               // f16 copies of inputs
  f16* xk16 = xq16 + MD;
  f16* xv16 = xk16 + MD;
  f16* wq16 = xv16 + MD;                              // f16 copies of weights
  f16* wk16 = wq16 + DD;
  f16* wv16 = wk16 + DD;
  f16* wo16 = wv16 + DD;

  // one-shot f32 -> f16 conversions (bandwidth bound)
  int cb_x = (int)(MD / 8 / 256);                     // 2048 blocks
  int cb_w = (int)(DD / 8 / 256);                     // 512 blocks
  cvt_kernel<<<cb_x, 256, 0, stream>>>(query, xq16, (int)MD);
  cvt_kernel<<<cb_x, 256, 0, stream>>>(key,   xk16, (int)MD);
  cvt_kernel<<<cb_x, 256, 0, stream>>>(value, xv16, (int)MD);
  cvt_kernel<<<cb_w, 256, 0, stream>>>(Wq, wq16, (int)DD);
  cvt_kernel<<<cb_w, 256, 0, stream>>>(Wk, wk16, (int)DD);
  cvt_kernel<<<cb_w, 256, 0, stream>>>(Wv, wv16, (int)DD);
  cvt_kernel<<<cb_w, 256, 0, stream>>>(Wo, wo16, (int)DD);

  dim3 gemm_grid(M_ / 16, D_ / 512);                  // 256 x 2 blocks, 8 waves ea.
  proj_kernel<<<gemm_grid, 256, 0, stream>>>(xq16, wq16, bq, q16);
  proj_kernel<<<gemm_grid, 256, 0, stream>>>(xk16, wk16, bk, k16);
  proj_kernel<<<gemm_grid, 256, 0, stream>>>(xv16, wv16, bv, v16);

  int attn_blocks = B_ * H_ * (S_ / 64);              // 1024 blocks, 4 waves each
  attn_kernel<<<attn_blocks, 128, 0, stream>>>(q16, k16, v16, o16);

  out_proj_kernel<<<gemm_grid, 256, 0, stream>>>(o16, wo16, bo, (float*)d_out);
}
